// GATModel_65300682768657
// MI455X (gfx1250) — matrix operand order repaired and checked by reading, compiled, not run on hardware
//
#include <hip/hip_runtime.h>
#include <hip/hip_bf16.h>
#include <math.h>

typedef __attribute__((ext_vector_type(16))) _Float16 v16h;
typedef __attribute__((ext_vector_type(8)))  float    v8f;

#define NUSR  512
#define INDIM 64
#define HIDD  128
#define NHEAD 4
#define HDIM  32
#define NANTS 16
#define NB    8

#define TSTRIDE 36   // u_tgt tile row stride in floats: 144B keeps rows 16B-aligned

static __device__ __forceinline__ float lrelu_f(float x) { return fmaxf(x, 0.2f * x); }

// ---------------------------------------------------------------------------
// Generic batched GEMM: C[z] = A[z/adiv] (MxK) @ W[z%wmod]^T (K x N), W is (N,K)
// row-major. One wave32 per 16x16 C tile, K in chunks of 32 via
// v_wmma_f32_16x16x32_f16 (f32 accumulate). Optional bias + lrelu epilogue.
// A/B operands built per CDNA5 16-bit WMMA VGPR layouts (ISA 7.12.2) with
// float2 (global_load_b64) gathers feeding v_cvt_pk_f16_f32.
// ---------------------------------------------------------------------------
__global__ void gemm_wmma_kernel(const float* __restrict__ A, int lda, long strideA, int adiv,
                                 const float* __restrict__ W, int ldw, long strideW, int wmod,
                                 float* __restrict__ C, int ldc, long strideC,
                                 const float* __restrict__ bias, int act, int K)
{
    const int lane  = threadIdx.x;       // 0..31
    const int hi    = lane >> 4;
    const int l15   = lane & 15;
    const int mbase = blockIdx.x * 16;
    const int nbase = blockIdx.y * 16;
    const int z     = blockIdx.z;

    const float* Ab = A + (long)(z / adiv) * strideA;
    const float* Wb = W + (long)(z % wmod) * strideW;
    float*       Cb = C + (long)z * strideC;

    v8f acc = {0.f, 0.f, 0.f, 0.f, 0.f, 0.f, 0.f, 0.f};

    for (int k0 = 0; k0 < K; k0 += 32) {
        union { v16h v; _Float16 e[16]; } a, b;
        // A 16x32 f16 layout: lane holds row M=l15; K slots depend on hi half.
        const float* arow = Ab + (long)(mbase + l15) * lda + k0;
#pragma unroll
        for (int r = 0; r < 8; ++r) {
            const int kb = ((r < 4) ? 2 * r : 2 * r + 8) + (hi ? 8 : 0);
            const float2 av = *(const float2*)&arow[kb];     // 8B-aligned (kb even)
            a.e[2 * r]     = (_Float16)av.x;
            a.e[2 * r + 1] = (_Float16)av.y;
        }
        // B 32x16 f16 layout: lane holds col N=l15; K slots match A (0..7,16..23 lo / 8..15,24..31 hi).
        const float* wrow = Wb + (long)(nbase + l15) * ldw + k0;   // B[k][n] = W[n][k]
#pragma unroll
        for (int r = 0; r < 8; ++r) {
            const int kb = ((r < 4) ? 2 * r : 2 * r + 8) + (hi ? 8 : 0);
            const float2 wv = *(const float2*)&wrow[kb];
            b.e[2 * r]     = (_Float16)wv.x;
            b.e[2 * r + 1] = (_Float16)wv.y;
        }
        acc = __builtin_amdgcn_wmma_f32_16x16x32_f16(false, a.v, false, b.v,
                                                     (short)0, acc, false, false);
    }

    // C layout: lane holds col N=l15; VGPR r is row M=r (lo lanes) / r+8 (hi lanes).
    const int n = nbase + l15;
    const int m0 = mbase + (hi ? 8 : 0);
    if (act == 1) {                       // uniform branch, bias load hoisted
        const float bv = bias[n];
#pragma unroll
        for (int r = 0; r < 8; ++r)
            Cb[(long)(m0 + r) * ldc + n] = lrelu_f(acc[r] + bv);
    } else {
#pragma unroll
        for (int r = 0; r < 8; ++r)
            Cb[(long)(m0 + r) * ldc + n] = acc[r];
    }
}

// ---------------------------------------------------------------------------
// Fused LayerNorm(32) + lrelu + attention-row-dot. One wave32 per row (HD==32).
// Stores u in place and dout[row] = dot_scale * <attn_k, u_row>.
// ---------------------------------------------------------------------------
__global__ void ln_lrelu_dot_kernel(float* __restrict__ U,
                                    const float* __restrict__ g,
                                    const float* __restrict__ be,
                                    const float* __restrict__ aw,
                                    float* __restrict__ dout,
                                    float dot_scale)
{
    const int lane = threadIdx.x & 31;
    const int row  = blockIdx.x * (blockDim.x >> 5) + (threadIdx.x >> 5);
    const int bk   = row >> 9;            // / NUSR
    const int k    = bk & (NHEAD - 1);

    float v = U[(long)row * HDIM + lane];
    float s = v;
#pragma unroll
    for (int m = 1; m < 32; m <<= 1) s += __shfl_xor(s, m);
    const float mean = s * (1.0f / 32.0f);
    const float d = v - mean;
    float q = d * d;
#pragma unroll
    for (int m = 1; m < 32; m <<= 1) q += __shfl_xor(q, m);
    const float var = q * (1.0f / 32.0f);
    const float y = d * rsqrtf(var + 1e-5f) * g[k * HDIM + lane] + be[k * HDIM + lane];
    const float u = lrelu_f(y);
    U[(long)row * HDIM + lane] = u;

    float t = u * aw[k * HDIM + lane];
#pragma unroll
    for (int m = 1; m < 32; m <<= 1) t += __shfl_xor(t, m);
    if (lane == 0) dout[row] = t * dot_scale;
}

// ---------------------------------------------------------------------------
// Fused attention: one wave32 handles 16 rows (i) for one (b,head).
// e = AS_i + AT_j + sum_h aw_h*|s_h + t_h|   (lrelu folded: lrelu(v)=0.6v+0.4|v|)
// p = exp(e) directly (|e| provably small -> no max subtraction), probabilities
// produced straight into the WMMA A-operand layout; P @ u_tgt runs as two
// v_wmma_f32_16x16x32_f16 per 32-j tile. u_tgt tiles are register
// double-buffered (b128 global loads) and read from LDS as b128; tile t+2
// is prefetched (global_prefetch_b8). Epilogue adds the residual in XOUT, relu.
// ---------------------------------------------------------------------------
__global__ void attn_fused_kernel(const float* __restrict__ Us_all,
                                  const float* __restrict__ Ut_all,
                                  const float* __restrict__ AS,
                                  const float* __restrict__ AT,
                                  const float* __restrict__ attn_w,
                                  float* XOUT)
{
    __shared__ __align__(16) float tT[32 * TSTRIDE];   // u_tgt tile [j][h]
    __shared__ float atT[32];

    const int lane  = threadIdx.x;
    const int hi    = lane >> 4;
    const int l15   = lane & 15;
    const int ibase = blockIdx.x * 16;
    const int bk    = blockIdx.y;
    const int b     = bk >> 2;
    const int k     = bk & 3;
    constexpr float kScale = 0.17677669529663687f;   // 1/sqrt(32)

    const float* Us  = Us_all + (long)bk * NUSR * HDIM;
    const float* Ut  = Ut_all + (long)bk * NUSR * HDIM;
    const float* asr = AS + (long)bk * NUSR;
    const float* atr = AT + (long)bk * NUSR;

    // per-lane row registers (float4 packed, fully unrolled use)
    const int i = ibase + l15;
    float4 s4[8], aw4[8];
#pragma unroll
    for (int c = 0; c < 8; ++c) s4[c] = *(const float4*)&Us[(long)i * HDIM + 4 * c];
#pragma unroll
    for (int c = 0; c < 8; ++c) {
        float4 a = *(const float4*)&attn_w[k * HDIM + 4 * c];
        const float f = 0.4f * kScale;
        aw4[c] = make_float4(a.x * f, a.y * f, a.z * f, a.w * f);
    }
    const float as_i = asr[i];

    v8f acc0 = {0.f, 0.f, 0.f, 0.f, 0.f, 0.f, 0.f, 0.f};
    v8f acc1 = {0.f, 0.f, 0.f, 0.f, 0.f, 0.f, 0.f, 0.f};
    float lsum = 0.0f;

    // preload tile 0 into registers (each lane owns row j = jt + lane)
    float4 tr[8];
    float  atreg;
    {
        const float* p = Ut + (long)lane * HDIM;
#pragma unroll
        for (int c = 0; c < 8; ++c) tr[c] = *(const float4*)&p[4 * c];
        atreg = atr[lane];
    }

    for (int jt = 0; jt < NUSR; jt += 32) {
        // commit register tile to LDS (ds_store_b128)
#pragma unroll
        for (int c = 0; c < 8; ++c)
            *(float4*)&tT[lane * TSTRIDE + 4 * c] = tr[c];
        atT[lane] = atreg;
        __syncthreads();

        // issue next tile's global loads now; the score loop below hides them
        if (jt + 32 < NUSR) {
            const float* p = Ut + (long)(jt + 32 + lane) * HDIM;
#pragma unroll
            for (int c = 0; c < 8; ++c) tr[c] = *(const float4*)&p[4 * c];
            atreg = atr[jt + 32 + lane];
        }
        if (jt + 64 < NUSR)
            __builtin_prefetch(Ut + (long)(jt + 64 + lane) * HDIM, 0, 3);

        // scores -> probabilities, directly in WMMA A (16x32 f16) lane layout
        union { v16h v; _Float16 e[16]; } pa;
        float lpart = 0.0f;
#pragma unroll
        for (int idx = 0; idx < 16; ++idx) {
            const int r  = idx >> 1;
            const int kb = ((r < 4) ? 2 * r : 2 * r + 8) + (hi ? 8 : 0);
            const int jl = kb + (idx & 1);
            const float4* trow = (const float4*)&tT[jl * TSTRIDE];
            float ev = as_i + atT[jl];
#pragma unroll
            for (int c = 0; c < 8; ++c) {
                const float4 tq = trow[c];          // ds_load_b128
                ev = fmaf(aw4[c].x, fabsf(s4[c].x + tq.x), ev);
                ev = fmaf(aw4[c].y, fabsf(s4[c].y + tq.y), ev);
                ev = fmaf(aw4[c].z, fabsf(s4[c].z + tq.z), ev);
                ev = fmaf(aw4[c].w, fabsf(s4[c].w + tq.w), ev);
            }
            const float p = __expf(ev);
            lpart += p;
            pa.e[idx] = (_Float16)p;
        }
        lsum += lpart;

        // B operands (K=j 32 x N=h 16), two N halves of u_tgt tile
        union { v16h v; _Float16 e[16]; } b0, b1;
#pragma unroll
        for (int r = 0; r < 8; ++r) {
            const int kb = ((r < 4) ? 2 * r : 2 * r + 8) + (hi ? 8 : 0);
            b0.e[2 * r]     = (_Float16)tT[kb * TSTRIDE + l15];
            b0.e[2 * r + 1] = (_Float16)tT[(kb + 1) * TSTRIDE + l15];
            b1.e[2 * r]     = (_Float16)tT[kb * TSTRIDE + l15 + 16];
            b1.e[2 * r + 1] = (_Float16)tT[(kb + 1) * TSTRIDE + l15 + 16];
        }
        acc0 = __builtin_amdgcn_wmma_f32_16x16x32_f16(false, pa.v, false, b0.v,
                                                      (short)0, acc0, false, false);
        acc1 = __builtin_amdgcn_wmma_f32_16x16x32_f16(false, pa.v, false, b1.v,
                                                      (short)0, acc1, false, false);
        __syncthreads();
    }

    // softmax denominator per row i: lanes L and L+16 jointly cover all j
    lsum += __shfl_xor(lsum, 16);
    const float linv = 1.0f / lsum;

#pragma unroll
    for (int r = 0; r < 8; ++r) {
        const int m = r + (hi ? 8 : 0);
        const float f = __shfl(linv, m);   // lane m holds l for row i=m
        const long rowoff = ((long)b * NUSR + (ibase + m)) * HIDD;
        const int c0 = k * HDIM + l15;
        const float v0 = acc0[r] * f + XOUT[rowoff + c0];
        const float v1 = acc1[r] * f + XOUT[rowoff + c0 + 16];
        XOUT[rowoff + c0]      = fmaxf(v0, 0.0f);
        XOUT[rowoff + c0 + 16] = fmaxf(v1, 0.0f);
    }
}

// ---------------------------------------------------------------------------
// Max pool over nodes: pooled[b][c] = max_i X[b][i][c], float4 columns.
// ---------------------------------------------------------------------------
__global__ void pool_kernel(const float* __restrict__ X, float* __restrict__ P)
{
    const int b  = blockIdx.x;
    const int c4 = threadIdx.x * 4;   // 32 threads x 4 cols
    float4 m = make_float4(-INFINITY, -INFINITY, -INFINITY, -INFINITY);
    for (int i = 0; i < NUSR; ++i) {
        const float4 v = *(const float4*)&X[((long)b * NUSR + i) * HIDD + c4];
        m.x = fmaxf(m.x, v.x); m.y = fmaxf(m.y, v.y);
        m.z = fmaxf(m.z, v.z); m.w = fmaxf(m.w, v.w);
    }
    *(float4*)&P[b * HIDD + c4] = m;
}

// ---------------------------------------------------------------------------
// Whole MLP tail for 8 rows in one block (negligible FLOPs).
// ---------------------------------------------------------------------------
__global__ void tail_kernel(const float* __restrict__ P,
                            const float* __restrict__ W1, const float* __restrict__ b1,
                            const float* __restrict__ W2, const float* __restrict__ b2,
                            const float* __restrict__ Wd1, const float* __restrict__ bd1,
                            const float* __restrict__ Wd2, const float* __restrict__ bd2,
                            const float* __restrict__ Wp1, const float* __restrict__ bp1,
                            const float* __restrict__ Wp2, const float* __restrict__ bp2,
                            float* __restrict__ out)
{
    __shared__ float h1[NB * HIDD];
    __shared__ float mlo[NB * 32];
    __shared__ float rd1[NB * 32];
    __shared__ float rp1[NB * 32];
    __shared__ float dls[NB * NANTS];
    __shared__ float pls[NB * NANTS];
    __shared__ float fd[NB];
    __shared__ float fp[NB];
    const int tid = threadIdx.x;   // 256 threads

    for (int idx = tid; idx < NB * HIDD; idx += 256) {
        const int b = idx >> 7, c = idx & 127;
        float acc = b1[c];
        for (int d = 0; d < HIDD; ++d) acc += P[b * HIDD + d] * W1[c * HIDD + d];
        h1[idx] = fmaxf(acc, 0.0f);
    }
    __syncthreads();

    {
        const int b = tid >> 5, o = tid & 31;
        float acc = b2[o];
        for (int d = 0; d < HIDD; ++d) acc += h1[b * HIDD + d] * W2[o * HIDD + d];
        mlo[tid] = fmaxf(acc, 0.0f);
    }
    __syncthreads();

    {
        const int b = tid >> 5, o = tid & 31;
        float accd = bd1[o], accp = bp1[o];
        for (int n = 0; n < NANTS; ++n) {
            accd += mlo[b * 32 + 16 + n] * Wd1[o * 16 + n];   // d_pred = mlp_out[:,16:]
            accp += mlo[b * 32 + n]      * Wp1[o * 16 + n];   // p_pred = mlp_out[:,:16]
        }
        rd1[tid] = fmaxf(accd, 0.0f);
        rp1[tid] = fmaxf(accp, 0.0f);
    }
    __syncthreads();

    if (tid < NB * NANTS) {
        const int b = tid >> 4, n = tid & 15;
        float accd = bd2[n], accp = bp2[n];
        for (int o = 0; o < 32; ++o) {
            accd += rd1[b * 32 + o] * Wd2[n * 32 + o];
            accp += rp1[b * 32 + o] * Wp2[n * 32 + o];
        }
        // raw_delta: clip[0,100] then relu (idempotent)
        float dv = fminf(fmaxf(accd, 0.0f), 100.0f);
        // raw_power: nan_to_num(nan->0, +inf->1, -inf->0) then relu
        float pv = accp;
        if (isnan(pv)) pv = 0.0f;
        else if (isinf(pv)) pv = (pv > 0.0f) ? 1.0f : 0.0f;
        pv = fmaxf(pv, 0.0f);
        dls[tid] = dv;
        pls[tid] = pv;
    }
    __syncthreads();

    if (tid < NB) {
        float sd = 0.0f, sp = 0.0f;
        for (int n = 0; n < NANTS; ++n) { sd += dls[tid * 16 + n]; sp += pls[tid * 16 + n]; }
        sd = fmaxf(sd, 1e-9f);
        sp = fmaxf(sp, 1e-9f);
        const float BMAX = 8.5f;          // 2*5.0 - 15*0.1
        fd[tid] = BMAX / fmaxf(BMAX, sd);
        fp[tid] = 1.0f / fmaxf(1.0f, sp); // PMAX = 1
    }
    __syncthreads();

    if (tid < NB * NANTS) {
        const int b = tid >> 4;
        out[tid] = pls[tid] * fp[b];                                          // power_scaled
        out[NB * NANTS + tid] = fminf(fmaxf(dls[tid] * fd[b], 0.0f), 100.0f); // delta_scaled
    }
}

// ---------------------------------------------------------------------------
extern "C" void kernel_launch(void* const* d_in, const int* in_sizes, int n_in,
                              void* d_out, int out_size, void* d_ws, size_t ws_size,
                              hipStream_t stream)
{
    (void)in_sizes; (void)n_in; (void)out_size; (void)ws_size;
    const float* x0 = (const float*)d_in[0];

    float* ws   = (float*)d_ws;
    float* XA   = ws;                                  // B*NU*HID
    float* XB   = XA + (long)NB * NUSR * HIDD;         // B*NU*HID
    float* USRC = XB + (long)NB * NUSR * HIDD;         // B*H*NU*HD
    float* UTGT = USRC + (long)NB * NHEAD * NUSR * HDIM;
    float* ASb  = UTGT + (long)NB * NHEAD * NUSR * HDIM;   // B*H*NU
    float* ATb  = ASb + (long)NB * NHEAD * NUSR;
    float* PO   = ATb + (long)NB * NHEAD * NUSR;           // B*HID

    constexpr float kScale = 0.17677669529663687f;
    const float dot_scale = 0.6f * kScale;

    const float* Xin = x0;
    int din = INDIM;
    float* XOUT = XA;

    for (int l = 0; l < 2; ++l) {
        const int pb = 1 + l * 9;
        const float* Wsrc = (const float*)d_in[pb + 0];
        const float* gsrc = (const float*)d_in[pb + 1];
        const float* bsrc = (const float*)d_in[pb + 2];
        const float* Wtgt = (const float*)d_in[pb + 3];
        const float* gtgt = (const float*)d_in[pb + 4];
        const float* btgt = (const float*)d_in[pb + 5];
        const float* attn = (const float*)d_in[pb + 6];
        const float* Wres = (const float*)d_in[pb + 7];
        const float* bres = (const float*)d_in[pb + 8];

        // u_src / u_tgt pre-activations: per (b,head) 512 x din x 32 GEMM
        dim3 gU(NUSR / 16, HDIM / 16, NB * NHEAD);
        gemm_wmma_kernel<<<gU, 32, 0, stream>>>(Xin, din, (long)NUSR * din, NHEAD,
                                                Wsrc, din, (long)HDIM * din, NHEAD,
                                                USRC, HDIM, (long)NUSR * HDIM,
                                                nullptr, 0, din);
        gemm_wmma_kernel<<<gU, 32, 0, stream>>>(Xin, din, (long)NUSR * din, NHEAD,
                                                Wtgt, din, (long)HDIM * din, NHEAD,
                                                UTGT, HDIM, (long)NUSR * HDIM,
                                                nullptr, 0, din);
        // residual: XOUT = lrelu(X @ Wres^T + bres)
        dim3 gR(NUSR / 16, HIDD / 16, NB);
        gemm_wmma_kernel<<<gR, 32, 0, stream>>>(Xin, din, (long)NUSR * din, 1,
                                                Wres, din, 0L, 1,
                                                XOUT, HIDD, (long)NUSR * HIDD,
                                                bres, 1, din);
        // LN + lrelu + row dots (0.6*scale folded in)
        const int rows = NB * NHEAD * NUSR;
        ln_lrelu_dot_kernel<<<rows / 8, 256, 0, stream>>>(USRC, gsrc, bsrc, attn, ASb, dot_scale);
        ln_lrelu_dot_kernel<<<rows / 8, 256, 0, stream>>>(UTGT, gtgt, btgt, attn, ATb, dot_scale);
        // fused attention (adds heads onto residual, applies relu)
        dim3 gA(NUSR / 16, NB * NHEAD);
        attn_fused_kernel<<<gA, 32, 0, stream>>>(USRC, UTGT, ASb, ATb, attn, XOUT);

        Xin = XOUT;
        din = HIDD;
        XOUT = XB;
    }

    pool_kernel<<<NB, 32, 0, stream>>>(XB, PO);
    tail_kernel<<<1, 256, 0, stream>>>(PO,
                                       (const float*)d_in[19], (const float*)d_in[20],
                                       (const float*)d_in[21], (const float*)d_in[22],
                                       (const float*)d_in[23], (const float*)d_in[24],
                                       (const float*)d_in[25], (const float*)d_in[26],
                                       (const float*)d_in[27], (const float*)d_in[28],
                                       (const float*)d_in[29], (const float*)d_in[30],
                                       (float*)d_out);
}
